// TransformerBlock_18657337934256
// MI455X (gfx1250) — matrix operand order repaired and checked by reading, not compiled, not run on hardware
//
#include <hip/hip_runtime.h>

typedef __bf16 bf16;
typedef __attribute__((ext_vector_type(16))) __bf16 bf16x16;
typedef __attribute__((ext_vector_type(8)))  float  f32x8;
typedef int v4i __attribute__((vector_size(16)));

#define D_    1024
#define H_    16
#define DH_   64
#define B_    4
#define N_    4096
#define WIN_  128
#define CH_   32            /* chunks = N/WIN */
#define ROWS_ (B_ * N_)     /* 16384 */

#if defined(__AMDGCN__) &&                                              \
    __has_builtin(__builtin_amdgcn_global_load_async_to_lds_b128) &&    \
    __has_builtin(__builtin_amdgcn_s_wait_asynccnt)
#define HAVE_ASYNC 1
#else
#define HAVE_ASYNC 0
#endif

// ---------------------------------------------------------------------------
// WMMA helpers (CDNA5 wave32, 16x16x32 bf16 -> f32)
// ---------------------------------------------------------------------------
__device__ __forceinline__ f32x8 wmma_bf(bf16x16 a, bf16x16 b, f32x8 c) {
  return __builtin_amdgcn_wmma_f32_16x16x32_bf16(false, a, false, b, (short)0, c,
                                                 false, false);
}

union FragU { bf16x16 v; uint4 q[2]; };

// A-fragment (16 rows x 32 K), row-major source with pitch (bf16 elems).
// Lane r=lane&15 holds row r; half=lane>>4 selects K chunks {h*8.., h*8+16..}.
__device__ __forceinline__ bf16x16 frag_a(const bf16* __restrict__ base, int pitch) {
  int lane = threadIdx.x & 31;
  int r = lane & 15, hf = lane >> 4;
  const bf16* p = base + (size_t)r * pitch + hf * 8;
  FragU f;
  f.q[0] = *(const uint4*)(p);
  f.q[1] = *(const uint4*)(p + 16);
  return f.v;
}

// B-fragment (32 K x 16 cols) from an N-major buffer baseT[n][k].
// Lane r holds column n=r; half selects K chunks {h*8.., h*8+16..}.
__device__ __forceinline__ bf16x16 frag_b(const bf16* __restrict__ baseT, int pitch) {
  int lane = threadIdx.x & 31;
  int r = lane & 15, hf = lane >> 4;
  const bf16* p = baseT + (size_t)r * pitch + hf * 8;
  FragU f;
  f.q[0] = *(const uint4*)(p);
  f.q[1] = *(const uint4*)(p + 16);
  return f.v;
}

// ---------------------------------------------------------------------------
// fp32 -> bf16 transposing conversion: src[K][Nn] -> dst[Nn][K]
// ---------------------------------------------------------------------------
__global__ void __launch_bounds__(256)
k_cvtT(const float* __restrict__ s, bf16* __restrict__ d, int K, int Nn) {
  __shared__ float tile[32][33];
  int n0 = blockIdx.x * 32, k0 = blockIdx.y * 32;
  int tx = threadIdx.x & 31, ty = threadIdx.x >> 5;  // 32 x 8
#pragma unroll
  for (int i = ty; i < 32; i += 8)
    tile[i][tx] = s[(size_t)(k0 + i) * Nn + n0 + tx];
  __syncthreads();
#pragma unroll
  for (int i = ty; i < 32; i += 8)
    d[(size_t)(n0 + i) * K + k0 + tx] = (bf16)tile[tx][i];
}

// ---------------------------------------------------------------------------
// LayerNorm over D=1024, one block (256 thr) per row, bf16 output
// ---------------------------------------------------------------------------
__global__ void __launch_bounds__(256)
k_layernorm(const float* __restrict__ x, const float* __restrict__ g,
            const float* __restrict__ bb, bf16* __restrict__ out) {
  int row = blockIdx.x;
  const float* xr = x + (size_t)row * D_;
  float v[4];
  float s = 0.f;
#pragma unroll
  for (int i = 0; i < 4; ++i) { v[i] = xr[threadIdx.x + i * 256]; s += v[i]; }
  __shared__ float red[8];
#pragma unroll
  for (int m = 16; m >= 1; m >>= 1) s += __shfl_xor(s, m, 32);
  if ((threadIdx.x & 31) == 0) red[threadIdx.x >> 5] = s;
  __syncthreads();
  float tot = 0.f;
#pragma unroll
  for (int i = 0; i < 8; ++i) tot += red[i];
  float mu = tot * (1.0f / D_);
  float ss = 0.f;
#pragma unroll
  for (int i = 0; i < 4; ++i) { float d0 = v[i] - mu; ss += d0 * d0; }
#pragma unroll
  for (int m = 16; m >= 1; m >>= 1) ss += __shfl_xor(ss, m, 32);
  __syncthreads();
  if ((threadIdx.x & 31) == 0) red[threadIdx.x >> 5] = ss;
  __syncthreads();
  float tot2 = 0.f;
#pragma unroll
  for (int i = 0; i < 8; ++i) tot2 += red[i];
  float rs = rsqrtf(tot2 * (1.0f / D_) + 1e-5f);
#pragma unroll
  for (int i = 0; i < 4; ++i) {
    int c = threadIdx.x + i * 256;
    out[(size_t)row * D_ + c] = (bf16)((v[i] - mu) * rs * g[c] + bb[c]);
  }
}

// ---------------------------------------------------------------------------
// Tiled WMMA GEMM: C[M,Nn] = A[M,K] * Wt[Nn,K]^T + bias, fused epilogue.
// Block tile 128x128, 256 threads = 8 waves, each wave a 32x64 subtile.
// Wt panel staged into double-buffered LDS with GLOBAL_LOAD_ASYNC_TO_LDS_B128
// when available (ASYNCcnt pipeline), else synchronous b128 copies.
// EPI 0: bias -> bf16; EPI 1: bias+GELU -> bf16; EPI 2: bias+resid -> f32.
// ---------------------------------------------------------------------------
template <int EPI>
__global__ void __launch_bounds__(256)
k_gemm(const bf16* __restrict__ A, const bf16* __restrict__ Wt,
       const float* __restrict__ bias, const float* __restrict__ resid,
       void* __restrict__ Cout, int M, int Nn, int K) {
  __shared__ bf16 WtS[2][128][32];  // [buf][n][k], rows are 64B
  int n0 = blockIdx.x * 128;
  int m0 = blockIdx.y * 128;
  int t = threadIdx.x;
  int wave = t >> 5, lane = t & 31, r = lane & 15, hf = lane >> 4;
  int wm = (wave >> 1) * 32, wn = (wave & 1) * 64;

  const bf16* wt_base = Wt + (size_t)n0 * K;

  // stage one 128n x 32k panel of Wt into LDS buffer `buf`
  auto stage = [&](int buf, int k0) {
#pragma unroll
    for (int j = 0; j < 2; ++j) {
      int cidx = t * 2 + j;           // 0..511 16-byte chunks
      int row = cidx >> 2;            // 0..127
      int qo = (cidx & 3) * 8;        // bf16 offset in row
      const bf16* src = wt_base + (size_t)row * K + k0 + qo;
      bf16* dst = &WtS[buf][row][qo];
#if HAVE_ASYNC
      __builtin_amdgcn_global_load_async_to_lds_b128((v4i*)src, (v4i*)dst, 0, 0);
#else
      *(uint4*)dst = *(const uint4*)src;
#endif
    }
  };

  f32x8 zero = {};
  f32x8 acc0[4] = {zero, zero, zero, zero};
  f32x8 acc1[4] = {zero, zero, zero, zero};

  int nk = K / 32;
  stage(0, 0);
  for (int ki = 0; ki < nk; ++ki) {
    int cur = ki & 1;
    if (ki + 1 < nk) stage(cur ^ 1, (ki + 1) * 32);
#if HAVE_ASYNC
    if (ki + 1 < nk)
      __builtin_amdgcn_s_wait_asynccnt(2);  // current panel landed
    else
      __builtin_amdgcn_s_wait_asynccnt(0);
#endif
    __syncthreads();
    int k0 = ki * 32;
    bf16x16 a0 = frag_a(A + (size_t)(m0 + wm) * K + k0, K);
    bf16x16 a1 = frag_a(A + (size_t)(m0 + wm + 16) * K + k0, K);
#pragma unroll
    for (int jn = 0; jn < 4; ++jn) {
      bf16x16 b = frag_b(&WtS[cur][wn + jn * 16][0], 32);
      acc0[jn] = wmma_bf(a0, b, acc0[jn]);
      acc1[jn] = wmma_bf(a1, b, acc1[jn]);
    }
    __syncthreads();
  }

#pragma unroll
  for (int im = 0; im < 2; ++im) {
#pragma unroll
    for (int jn = 0; jn < 4; ++jn) {
      f32x8 acc = im == 0 ? acc0[jn] : acc1[jn];
      int col = n0 + wn + jn * 16 + r;
      float bv = bias[col];
#pragma unroll
      for (int i = 0; i < 8; ++i) {
        int row = m0 + wm + im * 16 + hf * 8 + i;
        float val = acc[i] + bv;
        if (EPI == 1) val = 0.5f * val * (1.0f + erff(val * 0.70710678f));
        if (EPI == 2) {
          ((float*)Cout)[(size_t)row * Nn + col] =
              val + resid[(size_t)row * Nn + col];
        } else {
          ((bf16*)Cout)[(size_t)row * Nn + col] = (bf16)val;
        }
      }
    }
  }
}

// ---------------------------------------------------------------------------
// Landmarks: per (b,h,chunk) mean of k and v over the 128-token window
// ---------------------------------------------------------------------------
__global__ void __launch_bounds__(64)
k_landmark(const bf16* __restrict__ kmat, const bf16* __restrict__ vmat,
           bf16* __restrict__ klm, bf16* __restrict__ vlm) {
  int blk = blockIdx.x;  // b*H*C + h*C + c
  int c = blk & (CH_ - 1);
  int h = (blk / CH_) & (H_ - 1);
  int b = blk / (CH_ * H_);
  int d = threadIdx.x;
  size_t base = ((size_t)b * N_ + (size_t)c * WIN_) * D_ + h * DH_ + d;
  float sk = 0.f, sv = 0.f;
  for (int j = 0; j < WIN_; ++j) {
    sk += (float)kmat[base + (size_t)j * D_];
    sv += (float)vmat[base + (size_t)j * D_];
  }
  size_t o = (size_t)blk * DH_ + d;
  klm[o] = (bf16)(sk * (1.0f / WIN_));
  vlm[o] = (bf16)(sv * (1.0f / WIN_));
}

// ---------------------------------------------------------------------------
// EVA attention: one block per (b,h,chunk). 8 waves x 16 q-rows each.
// Scores (128 local + 32 global cols) live in registers; softmax via
// half-wave shfl_xor reductions; P goes through LDS (bf16) for the PV WMMAs.
// ---------------------------------------------------------------------------
__global__ void __launch_bounds__(256)
k_attention(const bf16* __restrict__ q, const bf16* __restrict__ kk,
            const bf16* __restrict__ v, const bf16* __restrict__ klm,
            const bf16* __restrict__ vlm, bf16* __restrict__ aout) {
  __shared__ bf16 vT[64][128];    // v transposed: [dh][j]
  __shared__ bf16 vlmT[64][32];   // v_lm transposed: [dh][e]
  __shared__ bf16 pS[128][160];   // softmaxed probs, bf16

  int blk = blockIdx.x;
  int c = blk & (CH_ - 1);
  int h = (blk / CH_) & (H_ - 1);
  int b = blk / (CH_ * H_);
  int baseRow = b * N_ + c * WIN_;
  int t = threadIdx.x;

  for (int idx = t; idx < WIN_ * DH_; idx += 256) {
    int j = idx >> 6, d = idx & 63;
    vT[d][j] = v[(size_t)(baseRow + j) * D_ + h * DH_ + d];
  }
  size_t lmBase = ((size_t)(b * H_ + h)) * CH_ * DH_;
  for (int idx = t; idx < CH_ * DH_; idx += 256) {
    int e = idx >> 6, d = idx & 63;
    vlmT[d][e] = vlm[lmBase + (size_t)e * DH_ + d];
  }
  __syncthreads();

  int wave = t >> 5, lane = t & 31, r = lane & 15, hf = lane >> 4;
  int m0 = wave * 16;

  const bf16* qbase = q + (size_t)(baseRow + m0) * D_ + h * DH_;
  bf16x16 qa0 = frag_a(qbase, D_);       // K = 0..31
  bf16x16 qa1 = frag_a(qbase + 32, D_);  // K = 32..63

  f32x8 zero = {};
  f32x8 sl[8];
  f32x8 sg[2];
#pragma unroll
  for (int jt = 0; jt < 8; ++jt) {
    f32x8 acc = zero;
    const bf16* kb = kk + (size_t)(baseRow + jt * 16) * D_ + h * DH_;
    acc = wmma_bf(qa0, frag_b(kb, D_), acc);
    acc = wmma_bf(qa1, frag_b(kb + 32, D_), acc);
    sl[jt] = acc;
  }
#pragma unroll
  for (int et = 0; et < 2; ++et) {
    f32x8 acc = zero;
    const bf16* gb = klm + lmBase + (size_t)(et * 16) * DH_;
    acc = wmma_bf(qa0, frag_b(gb, DH_), acc);
    acc = wmma_bf(qa1, frag_b(gb + 32, DH_), acc);
    sg[et] = acc;
  }

  const float scale = 0.125f;  // 1/sqrt(DH)
#pragma unroll
  for (int jt = 0; jt < 8; ++jt) sl[jt] *= scale;
#pragma unroll
  for (int et = 0; et < 2; ++et) {
    sg[et] *= scale;
    if (et * 16 + r == c) {
#pragma unroll
      for (int i = 0; i < 8; ++i) sg[et][i] = -1.0e9f;
    }
  }

  // softmax: element i of each accumulator is row (m0 + hf*8 + i); a full row
  // lives in the 16 lanes of this half-wave -> reduce with xor 1/2/4/8.
  float mx[8], sm[8];
#pragma unroll
  for (int i = 0; i < 8; ++i) {
    float m = sl[0][i];
#pragma unroll
    for (int jt = 1; jt < 8; ++jt) m = fmaxf(m, sl[jt][i]);
    m = fmaxf(m, fmaxf(sg[0][i], sg[1][i]));
#pragma unroll
    for (int o = 8; o >= 1; o >>= 1) m = fmaxf(m, __shfl_xor(m, o, 32));
    mx[i] = m;
  }
#pragma unroll
  for (int i = 0; i < 8; ++i) {
    float s = 0.f;
#pragma unroll
    for (int jt = 0; jt < 8; ++jt) {
      float e = __expf(sl[jt][i] - mx[i]);
      sl[jt][i] = e;
      s += e;
    }
    {
      float e = __expf(sg[0][i] - mx[i]); sg[0][i] = e; s += e;
      e = __expf(sg[1][i] - mx[i]);       sg[1][i] = e; s += e;
    }
#pragma unroll
    for (int o = 8; o >= 1; o >>= 1) s += __shfl_xor(s, o, 32);
    sm[i] = 1.0f / s;
  }

#pragma unroll
  for (int jt = 0; jt < 8; ++jt)
#pragma unroll
    for (int i = 0; i < 8; ++i)
      pS[m0 + hf * 8 + i][jt * 16 + r] = (bf16)(sl[jt][i] * sm[i]);
#pragma unroll
  for (int et = 0; et < 2; ++et)
#pragma unroll
    for (int i = 0; i < 8; ++i)
      pS[m0 + hf * 8 + i][128 + et * 16 + r] = (bf16)(sg[et][i] * sm[i]);
  __syncthreads();

  // out = P_loc(128x128) @ V(128x64) + P_glob(128x32) @ V_lm(32x64)
  f32x8 oc[4] = {zero, zero, zero, zero};
#pragma unroll
  for (int ks = 0; ks < 4; ++ks) {
    bf16x16 pa = frag_a(&pS[m0][ks * 32], 160);
#pragma unroll
    for (int dt = 0; dt < 4; ++dt)
      oc[dt] = wmma_bf(pa, frag_b(&vT[dt * 16][ks * 32], 128), oc[dt]);
  }
  {
    bf16x16 pg = frag_a(&pS[m0][128], 160);
#pragma unroll
    for (int dt = 0; dt < 4; ++dt)
      oc[dt] = wmma_bf(pg, frag_b(&vlmT[dt * 16][0], 32), oc[dt]);
  }

#pragma unroll
  for (int dt = 0; dt < 4; ++dt)
#pragma unroll
    for (int i = 0; i < 8; ++i)
      aout[(size_t)(baseRow + m0 + hf * 8 + i) * D_ + h * DH_ + dt * 16 + r] =
          (bf16)(oc[dt][i]);
}

// ---------------------------------------------------------------------------
// Host launcher
// ---------------------------------------------------------------------------
extern "C" void kernel_launch(void* const* d_in, const int* in_sizes, int n_in,
                              void* d_out, int out_size, void* d_ws,
                              size_t ws_size, hipStream_t stream) {
  const float* x   = (const float*)d_in[0];
  const float* Wq  = (const float*)d_in[1];
  const float* bq  = (const float*)d_in[2];
  const float* Wk  = (const float*)d_in[3];
  const float* bk  = (const float*)d_in[4];
  const float* Wv  = (const float*)d_in[5];
  const float* bv  = (const float*)d_in[6];
  const float* Wp  = (const float*)d_in[7];
  const float* bp  = (const float*)d_in[8];
  const float* g1  = (const float*)d_in[9];
  const float* be1 = (const float*)d_in[10];
  const float* g2  = (const float*)d_in[11];
  const float* be2 = (const float*)d_in[12];
  const float* W1  = (const float*)d_in[13];
  const float* b1f = (const float*)d_in[14];
  const float* W2  = (const float*)d_in[15];
  const float* b2f = (const float*)d_in[16];
  float* out = (float*)d_out;

  char* w = (char*)d_ws;
  auto alloc = [&](size_t bytes) -> char* {
    char* p = w;
    w += (bytes + 255) & ~(size_t)255;
    return p;
  };
  const size_t actB = (size_t)ROWS_ * D_;  // activation elems
  bf16* hb  = (bf16*)alloc(actB * 2);
  bf16* qb  = (bf16*)alloc(actB * 2);
  bf16* kb  = (bf16*)alloc(actB * 2);
  bf16* vb  = (bf16*)alloc(actB * 2);
  bf16* ab  = (bf16*)alloc(actB * 2);
  float* x1 = (float*)alloc(actB * 4);
  bf16* h2  = (bf16*)alloc(actB * 2);
  bf16* mid = (bf16*)alloc((size_t)ROWS_ * 4 * D_ * 2);
  bf16* WqT = (bf16*)alloc((size_t)D_ * D_ * 2);
  bf16* WkT = (bf16*)alloc((size_t)D_ * D_ * 2);
  bf16* WvT = (bf16*)alloc((size_t)D_ * D_ * 2);
  bf16* WpT = (bf16*)alloc((size_t)D_ * D_ * 2);
  bf16* W1T = (bf16*)alloc((size_t)D_ * 4 * D_ * 2);  // [4096][1024]
  bf16* W2T = (bf16*)alloc((size_t)4 * D_ * D_ * 2);  // [1024][4096]
  bf16* klm = (bf16*)alloc((size_t)B_ * H_ * CH_ * DH_ * 2);
  bf16* vlm = (bf16*)alloc((size_t)B_ * H_ * CH_ * DH_ * 2);

  // transpose+convert weights: W[K][N] -> Wt[N][K] bf16
  dim3 ct(D_ / 32, D_ / 32);
  hipLaunchKernelGGL(k_cvtT, ct, dim3(256), 0, stream, Wq, WqT, D_, D_);
  hipLaunchKernelGGL(k_cvtT, ct, dim3(256), 0, stream, Wk, WkT, D_, D_);
  hipLaunchKernelGGL(k_cvtT, ct, dim3(256), 0, stream, Wv, WvT, D_, D_);
  hipLaunchKernelGGL(k_cvtT, ct, dim3(256), 0, stream, Wp, WpT, D_, D_);
  hipLaunchKernelGGL(k_cvtT, dim3(4 * D_ / 32, D_ / 32), dim3(256), 0, stream,
                     W1, W1T, D_, 4 * D_);
  hipLaunchKernelGGL(k_cvtT, dim3(D_ / 32, 4 * D_ / 32), dim3(256), 0, stream,
                     W2, W2T, 4 * D_, D_);

  // h = LN(x)
  hipLaunchKernelGGL(k_layernorm, dim3(ROWS_), dim3(256), 0, stream, x, g1, be1, hb);

  // q/k/v = h @ W + b  (bf16 out)
  dim3 gproj(D_ / 128, ROWS_ / 128);
  hipLaunchKernelGGL((k_gemm<0>), gproj, dim3(256), 0, stream, hb, WqT, bq,
                     (const float*)nullptr, (void*)qb, ROWS_, D_, D_);
  hipLaunchKernelGGL((k_gemm<0>), gproj, dim3(256), 0, stream, hb, WkT, bk,
                     (const float*)nullptr, (void*)kb, ROWS_, D_, D_);
  hipLaunchKernelGGL((k_gemm<0>), gproj, dim3(256), 0, stream, hb, WvT, bv,
                     (const float*)nullptr, (void*)vb, ROWS_, D_, D_);

  // landmarks + attention
  hipLaunchKernelGGL(k_landmark, dim3(B_ * H_ * CH_), dim3(64), 0, stream, kb, vb,
                     klm, vlm);
  hipLaunchKernelGGL(k_attention, dim3(B_ * H_ * CH_), dim3(256), 0, stream, qb, kb,
                     vb, klm, vlm, ab);

  // x1 = x + a @ Wp + bp  (f32 out)
  hipLaunchKernelGGL((k_gemm<2>), gproj, dim3(256), 0, stream, ab, WpT, bp, x,
                     (void*)x1, ROWS_, D_, D_);

  // h2 = LN(x1)
  hipLaunchKernelGGL(k_layernorm, dim3(ROWS_), dim3(256), 0, stream, x1, g2, be2, h2);

  // mid = gelu(h2 @ W1 + b1)  (bf16 out)
  hipLaunchKernelGGL((k_gemm<1>), dim3(4 * D_ / 128, ROWS_ / 128), dim3(256), 0,
                     stream, h2, W1T, b1f, (const float*)nullptr, (void*)mid,
                     ROWS_, 4 * D_, D_);

  // out = x1 + mid @ W2 + b2  (f32 out)
  hipLaunchKernelGGL((k_gemm<2>), gproj, dim3(256), 0, stream, mid, W2T, b2f, x1,
                     (void*)out, ROWS_, D_, 4 * D_);
}
